// mutil_head_attention_50680614092917
// MI455X (gfx1250) — matrix operand order repaired and checked by reading, not compiled, not run on hardware
//
#include <hip/hip_runtime.h>
#include <hip/hip_bf16.h>
#include <math.h>

#define HEADS 8
#define DFC   96           // channel / K dim
#define BSZ   16
#define DIL   512
#define PIL   1024
#define OF    (DFC * HEADS) // 768

typedef __attribute__((ext_vector_type(16))) __bf16 bf16x16;
typedef __attribute__((ext_vector_type(8)))  float  f32x8;

union FragAB {
    bf16x16 v;
    uint4   q[2];
};
union Pack8 {
    __bf16 h[8];
    uint4  q;
};

__device__ __forceinline__ f32x8 wmma_bf16(bf16x16 a, bf16x16 b, f32x8 c) {
    return __builtin_amdgcn_wmma_f32_16x16x32_bf16(
        /*neg_a=*/false, a, /*neg_b=*/false, b,
        /*c_mod=*/(short)0, c, /*reuse_a=*/false, /*reuse_b=*/false);
}

// tanh via the hardware transcendental unit: tanh(x) = (e-1)/(e+1), e = exp(2x)
__device__ __forceinline__ float fast_tanh(float x) {
    float e = __expf(2.0f * x);
    return (e - 1.0f) / (e + 1.0f);
}

// LDS byte offset of a __shared__ pointer (flat->LDS aperture keeps low 32 bits)
__device__ __forceinline__ unsigned lds_off(const void* p) {
    return (unsigned)(unsigned long long)p;
}

// CDNA5 async bulk copy: global -> LDS, 16B per lane, tracked by ASYNCcnt.
__device__ __forceinline__ void async_copy_b128(unsigned lds_byte_off, const void* gptr) {
    asm volatile("global_load_async_to_lds_b128 %0, %1, off"
                 :
                 : "v"(lds_byte_off), "v"(gptr)
                 : "memory");
}
__device__ __forceinline__ void wait_async_zero() {
    asm volatile("s_wait_asynccnt 0x0" ::: "memory");
}

// Load A fragment (16x32 bf16, row-major MxK source rows of length DFC).
__device__ __forceinline__ bf16x16 load_a_frag(const __bf16* rowbase, int k0, int g) {
    FragAB f;
    const uint4* p = (const uint4*)(rowbase + k0 + g * 8);
    f.q[0] = p[0];      // K = k0 + g*8   .. +7
    f.q[1] = p[2];      // K = k0 + 16 + g*8 .. +7   (+16 elems = +2 uint4)
    return f.v;
}

// Load B fragment (32x16 bf16) from a transposed (NxK row-major) buffer.
__device__ __forceinline__ bf16x16 load_b_frag(const __bf16* rowbase, int k0, int g) {
    FragAB f;
    const uint4* p = (const uint4*)(rowbase + k0 + g * 8);
    f.q[0] = p[0];      // K = k0 + g*8 + 0..7 (2 per dword)
    f.q[1] = p[2];      // K = k0 + 16 + g*8 .. +7
    return f.v;
}

// ---------------------------------------------------------------------------
// Kernel 1: act[b, l, o] = relu( sum_c x[b, c, l] * W[o, c] + bias[o] ) -> bf16
// GEMM view per batch:  C(768 x L) = W(768x96) * x(96xL); stored transposed.
// grid: (L/64, OF/64, BSZ), block: 128 (4 waves); each wave owns a 32x32 tile.
// (staging does f32 -> bf16 conversion, so it uses the VGPR + ds_store path)
// ---------------------------------------------------------------------------
__global__ __launch_bounds__(128)
void act_gemm_kernel(const float* __restrict__ x, const float* __restrict__ W,
                     const float* __restrict__ bias, __bf16* __restrict__ act, int L) {
    __shared__ __bf16 Asm[64 * DFC];  // W tile rows (o) x c
    __shared__ __bf16 Bsm[64 * DFC];  // x tile transposed: rows (l) x c

    const int b  = blockIdx.z;
    const int l0 = blockIdx.x * 64;
    const int o0 = blockIdx.y * 64;
    const int tid = threadIdx.x;

    // Stage W tile (rows contiguous in K): f32 -> bf16
    for (int idx = tid; idx < 64 * DFC; idx += 128) {
        int o = idx / DFC, c = idx % DFC;
        Asm[idx] = (__bf16)W[(o0 + o) * DFC + c];
    }
    // Stage x tile transposed: Bsm[l][c] = x[b, c, l0+l] (coalesced along l)
    for (int idx = tid; idx < 64 * DFC; idx += 128) {
        int c = idx / 64, l = idx % 64;
        Bsm[l * DFC + c] = (__bf16)x[((size_t)b * DFC + c) * L + l0 + l];
    }
    __syncthreads();

    const int wave = tid >> 5, lane = tid & 31;
    const int wm = (wave >> 1) * 32;   // o offset of this wave
    const int wn = (wave & 1) * 32;    // l offset of this wave
    const int mn = lane & 15;
    const int g  = lane >> 4;

    f32x8 acc[2][2] = {};
#pragma unroll
    for (int k0 = 0; k0 < DFC; k0 += 32) {
        bf16x16 af[2], bf[2];
#pragma unroll
        for (int t = 0; t < 2; ++t) {
            af[t] = load_a_frag(Asm + (wm + t * 16 + mn) * DFC, k0, g);
            bf[t] = load_b_frag(Bsm + (wn + t * 16 + mn) * DFC, k0, g);
        }
#pragma unroll
        for (int i = 0; i < 2; ++i)
#pragma unroll
            for (int j = 0; j < 2; ++j)
                acc[i][j] = wmma_bf16(af[i], bf[j], acc[i][j]);
    }

    // Bias + relu + bf16 pack + store.  C layout: VGPR t -> M = t (lanes 0-15)
    // or M = 8+t (lanes 16-31); N = lane&15.  Per lane: 8 contiguous o values.
#pragma unroll
    for (int i = 0; i < 2; ++i) {
        const int orow = o0 + wm + i * 16 + g * 8;
        const float4 b0 = ((const float4*)(bias + orow))[0];
        const float4 b1 = ((const float4*)(bias + orow))[1];
        const float bv[8] = {b0.x, b0.y, b0.z, b0.w, b1.x, b1.y, b1.z, b1.w};
#pragma unroll
        for (int j = 0; j < 2; ++j) {
            const int l = l0 + wn + j * 16 + mn;
            Pack8 pk;
#pragma unroll
            for (int t = 0; t < 8; ++t) {
                float v = acc[i][j][t] + bv[t];
                v = v > 0.0f ? v : 0.0f;
                pk.h[t] = (__bf16)v;
            }
            *(uint4*)(act + ((size_t)b * L + l) * OF + orow) = pk.q;
        }
    }
}

// ---------------------------------------------------------------------------
// Kernel 2: imap tile = mean_h tanh( D_h(512x96) * P_h(1024x96)^T / sqrt(96) )
// then accumulate row sums (over p) and col sums (over i) with f32 atomics.
// Double-buffered CDNA5 async global->LDS pipeline: while WMMA consumes head
// h from buffer buf, the async engine (ASYNCcnt) fills buf^1 with head h+1;
// global_prefetch warms L2 for head h+2.
// grid: (PIL/64, DIL/64, BSZ), block 128 (4 waves, each a 32x32 tile).
// ---------------------------------------------------------------------------
__global__ __launch_bounds__(128)
void interact_kernel(const __bf16* __restrict__ Dact, const __bf16* __restrict__ Pact,
                     float* __restrict__ rowsum, float* __restrict__ colsum) {
    __shared__ __bf16 Dsm[2][64 * DFC];
    __shared__ __bf16 Psm[2][64 * DFC];

    const int b  = blockIdx.z;
    const int i0 = blockIdx.y * 64;
    const int p0 = blockIdx.x * 64;
    const int tid = threadIdx.x;
    const int wave = tid >> 5, lane = tid & 31;
    const int wm = (wave >> 1) * 32;   // i offset of wave
    const int wn = (wave & 1) * 32;    // p offset of wave
    const int mn = lane & 15;
    const int g  = lane >> 4;
    const float rscale = 0.102062072616f;  // 1/sqrt(96)
    const int NQ = (64 * DFC * 2) / 16;    // 768 uint4 per tile

    const unsigned sD[2] = {lds_off(Dsm[0]), lds_off(Dsm[1])};
    const unsigned sP[2] = {lds_off(Psm[0]), lds_off(Psm[1])};

    // Issue one head's D+P tiles into LDS buffer `buf` via async copies.
    auto issue_head = [&](int h, int buf) {
        const uint4* gD = (const uint4*)(Dact + (((size_t)b * HEADS + h) * DIL + i0) * DFC);
        const uint4* gP = (const uint4*)(Pact + (((size_t)b * HEADS + h) * PIL + p0) * DFC);
        for (int idx = tid; idx < NQ; idx += 128) {
            async_copy_b128(sD[buf] + (unsigned)idx * 16u, gD + idx);
            async_copy_b128(sP[buf] + (unsigned)idx * 16u, gP + idx);
        }
    };

    f32x8 mean[2][2] = {};

    issue_head(0, 0);                       // prologue: fill buffer 0
    for (int h = 0; h < HEADS; ++h) {
        const int buf = h & 1;
        wait_async_zero();                  // head h's tiles resident in LDS
        __syncthreads();                    // (also: buf^1 readers from h-1 done)
        if (h + 1 < HEADS) issue_head(h + 1, buf ^ 1);  // overlap with compute
        if (h + 2 < HEADS) {                // warm L2 one stage further ahead
            const char* nD = (const char*)(Dact + (((size_t)b * HEADS + h + 2) * DIL + i0) * DFC);
            const char* nP = (const char*)(Pact + (((size_t)b * HEADS + h + 2) * PIL + p0) * DFC);
            __builtin_prefetch(nD + tid * 96, 0, 1);
            __builtin_prefetch(nP + tid * 96, 0, 1);
        }

        f32x8 s[2][2] = {};
#pragma unroll
        for (int k0 = 0; k0 < DFC; k0 += 32) {
            bf16x16 af[2], bf[2];
#pragma unroll
            for (int t = 0; t < 2; ++t) {
                af[t] = load_a_frag(Dsm[buf] + (wm + t * 16 + mn) * DFC, k0, g);
                bf[t] = load_b_frag(Psm[buf] + (wn + t * 16 + mn) * DFC, k0, g);
            }
#pragma unroll
            for (int i = 0; i < 2; ++i)
#pragma unroll
                for (int j = 0; j < 2; ++j)
                    s[i][j] = wmma_bf16(af[i], bf[j], s[i][j]);
        }
#pragma unroll
        for (int i = 0; i < 2; ++i)
#pragma unroll
            for (int j = 0; j < 2; ++j)
#pragma unroll
                for (int t = 0; t < 8; ++t)
                    mean[i][j][t] += fast_tanh(s[i][j][t] * rscale);

        __syncthreads();                    // all waves done reading buf
    }

    // mean over heads
#pragma unroll
    for (int i = 0; i < 2; ++i)
#pragma unroll
        for (int j = 0; j < 2; ++j)
#pragma unroll
            for (int t = 0; t < 8; ++t)
                mean[i][j][t] *= 0.125f;

    // --- row sums (sum over p): reduce across the 16 lanes of each half ---
#pragma unroll
    for (int it = 0; it < 2; ++it) {
        float rs[8];
#pragma unroll
        for (int t = 0; t < 8; ++t) rs[t] = mean[it][0][t] + mean[it][1][t];
#pragma unroll
        for (int mask = 1; mask < 16; mask <<= 1)
#pragma unroll
            for (int t = 0; t < 8; ++t) rs[t] += __shfl_xor(rs[t], mask, 32);
        if (mn == 0) {
            const int irow = i0 + wm + it * 16 + g * 8;
#pragma unroll
            for (int t = 0; t < 8; ++t)
                atomicAdd(&rowsum[(size_t)b * DIL + irow + t], rs[t]);
        }
    }

    // --- col sums (sum over i): per-lane partial + pair-up lane L / L+16 ---
#pragma unroll
    for (int jt = 0; jt < 2; ++jt) {
        float cs = 0.0f;
#pragma unroll
        for (int it = 0; it < 2; ++it)
#pragma unroll
            for (int t = 0; t < 8; ++t) cs += mean[it][jt][t];
        cs += __shfl_xor(cs, 16, 32);
        if (g == 0)
            atomicAdd(&colsum[(size_t)b * PIL + p0 + wn + jt * 16 + mn], cs);
    }
}

// ---------------------------------------------------------------------------
// Kernel 3: out[b, c, l] = x[b, c, l] * tanh(sums[b, l])
// ---------------------------------------------------------------------------
__global__ __launch_bounds__(256)
void finalize_kernel(const float* __restrict__ x, const float* __restrict__ sums,
                     float* __restrict__ out, int L, int total) {
    int idx = blockIdx.x * blockDim.x + threadIdx.x;
    if (idx >= total) return;
    int l = idx % L;
    int bb = idx / (DFC * L);
    out[idx] = x[idx] * fast_tanh(sums[bb * L + l]);
}

extern "C" void kernel_launch(void* const* d_in, const int* in_sizes, int n_in,
                              void* d_out, int out_size, void* d_ws, size_t ws_size,
                              hipStream_t stream) {
    const float* drug    = (const float*)d_in[0];  // (16, 96, 512)
    const float* protein = (const float*)d_in[1];  // (16, 96, 1024)
    const float* Wd      = (const float*)d_in[2];  // (768, 96)
    const float* bd      = (const float*)d_in[3];  // (768)
    const float* Wp      = (const float*)d_in[4];  // (768, 96)
    const float* bp      = (const float*)d_in[5];  // (768)
    float* out = (float*)d_out;

    const size_t dact_elems = (size_t)BSZ * DIL * OF;   // 6,291,456
    const size_t pact_elems = (size_t)BSZ * PIL * OF;   // 12,582,912
    __bf16* Dact = (__bf16*)d_ws;
    __bf16* Pact = Dact + dact_elems;
    float* rowsum = (float*)(Pact + pact_elems);        // BSZ*DIL
    float* colsum = rowsum + (size_t)BSZ * DIL;         // BSZ*PIL

    hipMemsetAsync(rowsum, 0, (size_t)(BSZ * DIL + BSZ * PIL) * sizeof(float), stream);

    // Stage 1: activations (bf16)
    act_gemm_kernel<<<dim3(DIL / 64, OF / 64, BSZ), 128, 0, stream>>>(drug, Wd, bd, Dact, DIL);
    act_gemm_kernel<<<dim3(PIL / 64, OF / 64, BSZ), 128, 0, stream>>>(protein, Wp, bp, Pact, PIL);

    // Stage 2: interaction map + reductions (double-buffered async LDS pipeline)
    interact_kernel<<<dim3(PIL / 64, DIL / 64, BSZ), 128, 0, stream>>>(Dact, Pact, rowsum, colsum);

    // Stage 3: gated outputs (concatenated: drug part then protein part)
    const int n1 = BSZ * DFC * DIL;   // 786432
    const int n2 = BSZ * DFC * PIL;   // 1572864
    finalize_kernel<<<(n1 + 255) / 256, 256, 0, stream>>>(drug, rowsum, out, DIL, n1);
    finalize_kernel<<<(n2 + 255) / 256, 256, 0, stream>>>(protein, colsum, out + n1, PIL, n2);
}
